// CriticWithGNN_90391881711792
// MI455X (gfx1250) — compile-verified, matrix operand order checked by reading, not run on hardware
//
#include <hip/hip_runtime.h>

typedef unsigned short u16;
typedef __attribute__((ext_vector_type(16))) __bf16 v16bf;
typedef __attribute__((ext_vector_type(8)))  float  v8f;

#define N_NODES   50000
#define N_EDGES   800000
#define D_FEAT    64
#define N_AGENTS  1024
#define MSG_DIM   128
#define ETILE     32
#define NTILE     32

// ---------- helpers ----------
__device__ __forceinline__ u16 f2bf(float f) {
  return __builtin_bit_cast(u16, (__bf16)f);          // native f32->bf16 cvt
}
__device__ __forceinline__ float bf2f(u16 h) {
  return __uint_as_float(((unsigned int)h) << 16);
}
__device__ __forceinline__ unsigned int pack2bf(float a, float b) {
  return (unsigned int)__builtin_bit_cast(u16, (__bf16)a) |
         ((unsigned int)__builtin_bit_cast(u16, (__bf16)b) << 16);
}

// C layout: VGPR j, lanes 0-15 -> M=j, lanes 16-31 -> M=j+8; col = lane&15.
__device__ __forceinline__ void store_tile(u16* __restrict__ C, int ldc, v8f c,
                                           int lane, bool relu) {
  const int col = lane & 15;
  const int rb  = (lane >> 4) << 3;
#pragma unroll
  for (int j = 0; j < 8; ++j) {
    float v = c[j];
    if (relu) v = fmaxf(v, 0.f);                      // v_max_num_f32
    C[(rb + j) * ldc + col] = f2bf(v);
  }
}

// Dual-accumulator tiled GEMM: 32 LDS rows x global W^T [NT*16][K] -> LDS bf16.
// Per nt: one B fragment feeds two WMMAs (rows 0-15 and 16-31) -> B loaded once,
// and the two independent WMMAs can co-execute / hide XDL latency.
template <int KS>
__device__ __forceinline__ void gemm2_lds(const u16* __restrict__ A, int lda,
                                          const u16* __restrict__ Wt, int ldk,
                                          const float* __restrict__ bias,
                                          u16* __restrict__ C, int ldc,
                                          int ntiles, bool relu,
                                          int wave, int lane) {
  const int r  = lane & 15;
  const int kh = (lane >> 4) << 4;
  for (int nt = wave; nt < ntiles; nt += 8) {
    const float bv = bias[nt * 16 + r];
    v8f c0, c1;
#pragma unroll
    for (int j = 0; j < 8; ++j) { c0[j] = bv; c1[j] = bv; }
    const u16* Bp = Wt + nt * 16 * ldk + r * ldk + kh;
    __builtin_prefetch(Bp, 0, 3);
#pragma unroll
    for (int ks = 0; ks < KS; ++ks) {
      v16bf b  = *(const v16bf*)(Bp + ks * 32);
      v16bf a0 = *(const v16bf*)(A + r * lda + ks * 32 + kh);
      v16bf a1 = *(const v16bf*)(A + (16 + r) * lda + ks * 32 + kh);
      c0 = __builtin_amdgcn_wmma_f32_16x16x32_bf16(false, a0, false, b, (short)0,
                                                   c0, false, false);
      c1 = __builtin_amdgcn_wmma_f32_16x16x32_bf16(false, a1, false, b, (short)0,
                                                   c1, false, false);
    }
    store_tile(C + nt * 16, ldc, c0, lane, relu);
    store_tile(C + 16 * ldc + nt * 16, ldc, c1, lane, relu);
  }
}

// ---------- prep kernels ----------
__global__ void convt_kernel(const float* __restrict__ src, u16* __restrict__ dst,
                             int K, int N, int Kp) {
  int i = blockIdx.x * blockDim.x + threadIdx.x;
  if (i < N * Kp) {
    int n = i / Kp, k = i - n * Kp;
    float v = (k < K) ? src[k * N + n] : 0.f;   // transpose [K,N] -> [N,Kp], zero-pad K
    dst[i] = f2bf(v);
  }
}

__global__ void zero_kernel(float* __restrict__ aggr, int n, int* __restrict__ cnt) {
  int i = blockIdx.x * blockDim.x + threadIdx.x;
  if (i < n) aggr[i] = 0.f;
  if (i == 0) *cnt = 0;
}

// keep only edges whose receiver is an agent node (< N_AGENTS): only those
// can influence q[:n_agents] (exact dead-code elimination, ~50x less edge work)
__global__ void compact_kernel(const int* __restrict__ snd, const int* __restrict__ rcv,
                               int E, int2* __restrict__ out, int* __restrict__ cnt) {
  int e = blockIdx.x * blockDim.x + threadIdx.x;
  if (e < E) {
    int r = rcv[e];
    if (r < N_AGENTS) {
      int p = atomicAdd(cnt, 1);
      out[p] = make_int2(snd[e], r);
    }
  }
}

// ---------- edge message MLP + scatter ----------
__global__ __launch_bounds__(256) void edge_msg_kernel(
    const float* __restrict__ x, const int2* __restrict__ epairs,
    const int* __restrict__ ecnt,
    const u16* __restrict__ Wm1t, const u16* __restrict__ Wm2t,
    const u16* __restrict__ Wm3t,
    const float* __restrict__ bm1, const float* __restrict__ bm2,
    const float* __restrict__ bm3,
    float* __restrict__ aggr) {
  __shared__ __align__(32) u16 sIn[ETILE * 128];
  __shared__ __align__(32) u16 h1[ETILE * 256];
  __shared__ __align__(32) u16 h2[ETILE * 256];
  __shared__ int sSnd[ETILE];
  __shared__ int sRcv[ETILE];

  const int tid = threadIdx.x;
  const int wave = tid >> 5, lane = tid & 31;
  const int cnt = *ecnt;
  const int ntiles = (cnt + ETILE - 1) / ETILE;

  for (int tile = blockIdx.x; tile < ntiles; tile += gridDim.x) {
    const int ebase = tile * ETILE;
    if (tid < ETILE) {
      int idx = ebase + tid;
      if (idx < cnt) { int2 p = epairs[idx]; sSnd[tid] = p.x; sRcv[tid] = p.y; }
      else           { sSnd[tid] = -1;       sRcv[tid] = -1; }
    }
    __syncthreads();
    // gather concat(x[s], x[r]) -> bf16 LDS [ETILE,128]; float4 at a time
    for (int i = tid; i < ETILE * 32; i += 256) {
      int e = i >> 5, c4 = (i & 31) << 2;               // c4 in {0,4,...,124}
      int node = (c4 < 64) ? sSnd[e] : sRcv[e];
      float4 v = make_float4(0.f, 0.f, 0.f, 0.f);
      if (node >= 0) v = *(const float4*)(x + node * D_FEAT + (c4 & 63));
      *(uint2*)(sIn + e * 128 + c4) = make_uint2(pack2bf(v.x, v.y), pack2bf(v.z, v.w));
    }
    __syncthreads();
    // L1: [32,128] @ [128,256] relu
    gemm2_lds<4>(sIn, 128, Wm1t, 128, bm1, h1, 256, 16, true, wave, lane);
    __syncthreads();
    // L2: [32,256] @ [256,256] relu
    gemm2_lds<8>(h1, 256, Wm2t, 256, bm2, h2, 256, 16, true, wave, lane);
    __syncthreads();
    // L3: [32,256] @ [256,128] linear -> scatter-add to aggr[receiver]
    {
      const int r  = lane & 15;
      const int kh = (lane >> 4) << 4;
      const int nt = wave;                               // 8 waves x 8 col-tiles
      const float bv = bm3[nt * 16 + r];
      v8f c0, c1;
#pragma unroll
      for (int j = 0; j < 8; ++j) { c0[j] = bv; c1[j] = bv; }
      const u16* Bp = Wm3t + nt * 16 * 256 + r * 256 + kh;
#pragma unroll
      for (int ks = 0; ks < 8; ++ks) {
        v16bf b  = *(const v16bf*)(Bp + ks * 32);
        v16bf a0 = *(const v16bf*)(h2 + r * 256 + ks * 32 + kh);
        v16bf a1 = *(const v16bf*)(h2 + (16 + r) * 256 + ks * 32 + kh);
        c0 = __builtin_amdgcn_wmma_f32_16x16x32_bf16(false, a0, false, b, (short)0,
                                                     c0, false, false);
        c1 = __builtin_amdgcn_wmma_f32_16x16x32_bf16(false, a1, false, b, (short)0,
                                                     c1, false, false);
      }
      const int ncol = nt * 16 + r;
      const int rb = (lane >> 4) << 3;
#pragma unroll
      for (int j = 0; j < 8; ++j) {
        int r0 = sRcv[rb + j];
        if (r0 >= 0) atomicAdd(&aggr[r0 * MSG_DIM + ncol], c0[j]);
        int r1 = sRcv[16 + rb + j];
        if (r1 >= 0) atomicAdd(&aggr[r1 * MSG_DIM + ncol], c1[j]);
      }
    }
    __syncthreads();
  }
}

// ---------- fused aggregation MLP + node update + critic head (1024 agents) ----------
__global__ __launch_bounds__(256) void node_critic_kernel(
    const float* __restrict__ x, const float* __restrict__ actions,
    const float* __restrict__ aggr,
    const u16* __restrict__ Wa1t, const u16* __restrict__ Wa2t,
    const u16* __restrict__ Wa3t,
    const float* __restrict__ ba1, const float* __restrict__ ba2,
    const float* __restrict__ ba3,
    const u16* __restrict__ Wu1t, const u16* __restrict__ Wu2t,
    const u16* __restrict__ Wu3t,
    const float* __restrict__ bu1, const float* __restrict__ bu2,
    const float* __restrict__ bu3,
    const u16* __restrict__ Wactt, const float* __restrict__ bact,
    const u16* __restrict__ Wh1t, const u16* __restrict__ Wh2t,
    const float* __restrict__ bh1, const float* __restrict__ bh2,
    const float* __restrict__ Wq, const float* __restrict__ bq,
    float* __restrict__ out) {
  __shared__ __align__(32) u16 bufA[NTILE * 256];
  __shared__ __align__(32) u16 bufB[NTILE * 256];
  __shared__ __align__(32) u16 actS[NTILE * 32];

  const int tid = threadIdx.x;
  const int wave = tid >> 5, lane = tid & 31;
  const int nbase = blockIdx.x * NTILE;

  // stage aggr rows as bf16 [32,128]; float4 at a time
  for (int i = tid; i < NTILE * 32; i += 256) {
    int rrow = i >> 5, c4 = (i & 31) << 2;
    float4 v = *(const float4*)(aggr + (nbase + rrow) * MSG_DIM + c4);
    *(uint2*)(bufA + rrow * 128 + c4) = make_uint2(pack2bf(v.x, v.y), pack2bf(v.z, v.w));
  }
  __syncthreads();
  // aggregation MLP 128->128->128->128
  gemm2_lds<4>(bufA, 128, Wa1t, 128, ba1, bufB, 128, 8, true,  wave, lane); __syncthreads();
  gemm2_lds<4>(bufB, 128, Wa2t, 128, ba2, bufA, 128, 8, true,  wave, lane); __syncthreads();
  gemm2_lds<4>(bufA, 128, Wa3t, 128, ba3, bufB, 128, 8, false, wave, lane); __syncthreads();
  // u = concat(x_bf16[:,0:64], a[:,0:128]) -> bufA stride 192
  for (int i = tid; i < NTILE * 16; i += 256) {         // x part, float4
    int rrow = i >> 4, c4 = (i & 15) << 2;
    float4 v = *(const float4*)(x + (nbase + rrow) * D_FEAT + c4);
    *(uint2*)(bufA + rrow * 192 + c4) = make_uint2(pack2bf(v.x, v.y), pack2bf(v.z, v.w));
  }
  for (int i = tid; i < NTILE * 32; i += 256) {         // a part, 4 bf16 copies
    int rrow = i >> 5, c4 = (i & 31) << 2;
    *(uint2*)(bufA + rrow * 192 + 64 + c4) = *(const uint2*)(bufB + rrow * 128 + c4);
  }
  __syncthreads();
  // update MLP 192->256->256->128
  gemm2_lds<6>(bufA, 192, Wu1t, 192, bu1, bufB, 256, 16, true, wave, lane); __syncthreads();
  gemm2_lds<8>(bufB, 256, Wu2t, 256, bu2, bufA, 256, 16, true, wave, lane); __syncthreads();
  // feats -> bufB cols 0..127 of z
  gemm2_lds<8>(bufA, 256, Wu3t, 256, bu3, bufB, 256, 8, false, wave, lane);
  // stage actions, K padded 16->32 with zeros
  for (int i = tid; i < NTILE * 32; i += 256) {
    int rrow = i >> 5, c = i & 31;
    actS[i] = (c < 16) ? f2bf(actions[(nbase + rrow) * 16 + c]) : (u16)0;
  }
  __syncthreads();
  // ap = relu(actions @ W_act) -> bufB cols 128..255 of z
  gemm2_lds<1>(actS, 32, Wactt, 32, bact, bufB + 128, 256, 8, true, wave, lane);
  __syncthreads();
  // head 256->256->256
  gemm2_lds<8>(bufB, 256, Wh1t, 256, bh1, bufA, 256, 16, true, wave, lane); __syncthreads();
  gemm2_lds<8>(bufA, 256, Wh2t, 256, bh2, bufB, 256, 16, true, wave, lane); __syncthreads();
  // q = z @ W_q + b_q (GEMV, one agent per lane)
  if (tid < NTILE) {
    float s = bq[0];
    for (int c = 0; c < 256; ++c) s += bf2f(bufB[tid * 256 + c]) * Wq[c];
    out[nbase + tid] = s;
  }
}

// ---------- launch ----------
extern "C" void kernel_launch(void* const* d_in, const int* in_sizes, int n_in,
                              void* d_out, int out_size, void* d_ws, size_t ws_size,
                              hipStream_t stream) {
  const float* x        = (const float*)d_in[0];
  const float* actions  = (const float*)d_in[1];
  const int*   senders  = (const int*)d_in[2];
  const int*   receivers= (const int*)d_in[3];
  const float* W_m1 = (const float*)d_in[5];  const float* b_m1 = (const float*)d_in[6];
  const float* W_m2 = (const float*)d_in[7];  const float* b_m2 = (const float*)d_in[8];
  const float* W_m3 = (const float*)d_in[9];  const float* b_m3 = (const float*)d_in[10];
  const float* W_a1 = (const float*)d_in[11]; const float* b_a1 = (const float*)d_in[12];
  const float* W_a2 = (const float*)d_in[13]; const float* b_a2 = (const float*)d_in[14];
  const float* W_a3 = (const float*)d_in[15]; const float* b_a3 = (const float*)d_in[16];
  const float* W_u1 = (const float*)d_in[17]; const float* b_u1 = (const float*)d_in[18];
  const float* W_u2 = (const float*)d_in[19]; const float* b_u2 = (const float*)d_in[20];
  const float* W_u3 = (const float*)d_in[21]; const float* b_u3 = (const float*)d_in[22];
  const float* W_act= (const float*)d_in[23]; const float* b_act= (const float*)d_in[24];
  const float* W_h1 = (const float*)d_in[25]; const float* b_h1 = (const float*)d_in[26];
  const float* W_h2 = (const float*)d_in[27]; const float* b_h2 = (const float*)d_in[28];
  const float* W_q  = (const float*)d_in[29]; const float* b_q  = (const float*)d_in[30];

  char* ws = (char*)d_ws;
  size_t o = 0;
  auto take = [&](size_t bytes) { char* p = ws + o; o = (o + bytes + 255) & ~(size_t)255; return p; };
  u16* Wm1t = (u16*)take(256 * 128 * 2);
  u16* Wm2t = (u16*)take(256 * 256 * 2);
  u16* Wm3t = (u16*)take(128 * 256 * 2);
  u16* Wa1t = (u16*)take(128 * 128 * 2);
  u16* Wa2t = (u16*)take(128 * 128 * 2);
  u16* Wa3t = (u16*)take(128 * 128 * 2);
  u16* Wu1t = (u16*)take(256 * 192 * 2);
  u16* Wu2t = (u16*)take(256 * 256 * 2);
  u16* Wu3t = (u16*)take(128 * 256 * 2);
  u16* Watt = (u16*)take(128 * 32 * 2);
  u16* Wh1t = (u16*)take(256 * 256 * 2);
  u16* Wh2t = (u16*)take(256 * 256 * 2);
  int*   cnt    = (int*)take(4);
  float* aggr   = (float*)take((size_t)N_AGENTS * MSG_DIM * 4);
  int2*  epairs = (int2*)take((size_t)N_EDGES * 8);

  auto conv = [&](const float* src, u16* dst, int K, int N, int Kp) {
    int n = N * Kp;
    convt_kernel<<<(n + 255) / 256, 256, 0, stream>>>(src, dst, K, N, Kp);
  };
  conv(W_m1, Wm1t, 128, 256, 128);
  conv(W_m2, Wm2t, 256, 256, 256);
  conv(W_m3, Wm3t, 256, 128, 256);
  conv(W_a1, Wa1t, 128, 128, 128);
  conv(W_a2, Wa2t, 128, 128, 128);
  conv(W_a3, Wa3t, 128, 128, 128);
  conv(W_u1, Wu1t, 192, 256, 192);
  conv(W_u2, Wu2t, 256, 256, 256);
  conv(W_u3, Wu3t, 256, 128, 256);
  conv(W_act, Watt, 16, 128, 32);
  conv(W_h1, Wh1t, 256, 256, 256);
  conv(W_h2, Wh2t, 256, 256, 256);

  zero_kernel<<<(N_AGENTS * MSG_DIM + 255) / 256, 256, 0, stream>>>(aggr, N_AGENTS * MSG_DIM, cnt);
  compact_kernel<<<(N_EDGES + 255) / 256, 256, 0, stream>>>(senders, receivers, N_EDGES, epairs, cnt);
  edge_msg_kernel<<<640, 256, 0, stream>>>(x, epairs, cnt, Wm1t, Wm2t, Wm3t,
                                           b_m1, b_m2, b_m3, aggr);
  node_critic_kernel<<<N_AGENTS / NTILE, 256, 0, stream>>>(
      x, actions, aggr, Wa1t, Wa2t, Wa3t, b_a1, b_a2, b_a3,
      Wu1t, Wu2t, Wu3t, b_u1, b_u2, b_u3, Watt, b_act,
      Wh1t, Wh2t, b_h1, b_h2, W_q, b_q, (float*)d_out);
}
